// LogRegBaseline_90958817395132
// MI455X (gfx1250) — matrix operand order repaired and checked by reading, not compiled, not run on hardware
//
#include <hip/hip_runtime.h>
#include <hip/hip_bf16.h>

// MI455X (gfx1250) implementation.
// out[b,t,u] = mask(t < len[b]-1) * ( sum_v prefixOR(codes)[b,t,v] * W[u,v] + bias[u] )
//
// Strategy: prefix-OR -> f16 A fragments (exact, values are 0/1);
// W split into f16 hi+lo fragments (exact to f32 rounding when combined with
// f32-accumulating v_wmma_f32_16x16x32_f16). GEMM M=131072(padded) N=512 K=512.
// Length masking exploited: prefix pass stops at len[b]-1, GEMM waves whose
// entire 32-row group is masked skip the K-loop (halves HBM/L2 traffic on avg).

typedef __attribute__((ext_vector_type(16))) _Float16 v16h;
typedef __attribute__((ext_vector_type(8)))  float    v8f;

#define BATCH 64
#define TLEN  2048           // padded rows per batch (power of two)
#define TM1   2047           // real output rows per batch
#define VOC   512
#define KCH   (VOC / 32)     // 16 K-chunks of 32

// Fragment storage: [tile][kchunk][lane 0..31][16 halves], 32B per lane,
// 1KB per fragment, contiguous per (tile,kchunk). Offset in halves.
__device__ __forceinline__ size_t frag_off(int tile, int kc, int lane) {
  return (((size_t)tile * KCH + kc) * 32 + lane) * 16;
}

// ---------------------------------------------------------------------------
// Kernel 1: prefix-OR over time, written directly in WMMA A-fragment layout.
// A-matrix 16x32 f16 layout (ISA 7.12.2):
//   lanes 0-15  (M=lane):    halves 0..7 -> K=0..7,   halves 8..15 -> K=16..23
//   lanes 16-31 (M=lane-16): halves 0..7 -> K=8..15,  halves 8..15 -> K=24..31
// Only rows t < len[b]-1 are ever consumed by the GEMM, so stop there.
// ---------------------------------------------------------------------------
__global__ __launch_bounds__(256) void prefix_kernel(const float* __restrict__ codes,
                                                     const int* __restrict__ lengths,
                                                     _Float16* __restrict__ A) {
  const int b = blockIdx.x;
  const int v = blockIdx.y * blockDim.x + threadIdx.x;   // 0..511
  const int kc = v >> 5;
  const int ko = v & 31;
  const int lane_off = ((ko >> 3) & 1) << 4;             // K bit3 selects lane group
  const int h = (ko & 7) + (((ko >> 4) & 1) << 3);       // K bit4 selects half group
  const int tmax = min(TM1, lengths[b] - 1);             // block-uniform bound
  const float* src = codes + (size_t)b * TLEN * VOC + v;
  float run = 0.0f;
  for (int t = 0; t < tmax; ++t) {
    run = fmaxf(run, src[(size_t)t * VOC]);              // codes in {0,1}: OR == max
    const int m = b * TLEN + t;                          // padded row index
    A[frag_off(m >> 4, kc, (m & 15) + lane_off) + h] = (_Float16)run;
  }
}

// ---------------------------------------------------------------------------
// Kernel 2: W[u][v] (f32) -> hi/lo f16 in WMMA B-fragment layout.
// B-matrix 32x16 f16 layout: lane = K bit3 * 16 + N, halves = (K&7) + K bit4 * 8.
// Contraction is over v (=> B[k=v][n=u] = W[u][v]).
// ---------------------------------------------------------------------------
__global__ __launch_bounds__(256) void wswz_kernel(const float* __restrict__ W,
                                                   _Float16* __restrict__ Whi,
                                                   _Float16* __restrict__ Wlo) {
  const int idx = blockIdx.x * blockDim.x + threadIdx.x;  // VOC*VOC threads
  const int u = idx >> 9;
  const int v = idx & (VOC - 1);
  const float w  = W[idx];
  const _Float16 hi = (_Float16)w;
  const _Float16 lo = (_Float16)(w - (float)hi);
  const int lane = (((v >> 3) & 1) << 4) + (u & 15);
  const size_t off = frag_off(u >> 4, v >> 5, lane) + ((v & 7) + (((v >> 4) & 1) << 3));
  Whi[off] = hi;
  Wlo[off] = lo;
}

__device__ __forceinline__ v16h load_frag(const _Float16* __restrict__ p,
                                          int tile, int kc, int lane) {
  return *(const v16h*)(p + frag_off(tile, kc, lane));
}

__device__ __forceinline__ v8f wmma_f16(v16h a, v16h b, v8f c) {
  return __builtin_amdgcn_wmma_f32_16x16x32_f16(false, a, false, b,
                                                (short)0, c, false, false);
}

// ---------------------------------------------------------------------------
// Kernel 3: GEMM. Grid (M/128, N/128); block = 256 threads = 8 waves (4m x 2n).
// Each wave: 32 rows x 64 cols = 2x4 accumulators, K in 16 chunks of 32,
// two chained WMMAs (lo then hi) per accumulator per chunk.
// A 128-row block lies inside one batch (2048 % 128 == 0), so the mask limit
// is uniform per block; waves whose 32 rows are all masked skip the K-loop.
// ---------------------------------------------------------------------------
__global__ __launch_bounds__(256) void gemm_kernel(const _Float16* __restrict__ A,
                                                   const _Float16* __restrict__ Bhi,
                                                   const _Float16* __restrict__ Blo,
                                                   const float* __restrict__ bias,
                                                   const int* __restrict__ lengths,
                                                   float* __restrict__ out) {
  const int lane = threadIdx.x & 31;
  const int wave = threadIdx.x >> 5;
  const int wm = wave >> 1;                 // 0..3
  const int wn = wave & 1;                  // 0..1
  const int mt0 = blockIdx.x * 8 + wm * 2;  // base M-subtile (16-row units)
  const int nt0 = blockIdx.y * 8 + wn * 4;  // base N-subtile (16-col units)

  const int bb    = (mt0 * 16) >> 11;       // batch (uniform per block)
  const int limit = lengths[bb] - 1;        // valid rows: t < limit
  const int twave = (mt0 * 16) & (TLEN - 1);

  v8f acc[2][4] = {};

  if (twave < limit) {                      // any valid row in this wave's group?
    for (int kc = 0; kc < KCH; ++kc) {
      const v16h a0 = load_frag(A, mt0 + 0, kc, lane);
      const v16h a1 = load_frag(A, mt0 + 1, kc, lane);
#pragma unroll
      for (int nt = 0; nt < 4; ++nt) {
        const v16h bl = load_frag(Blo, nt0 + nt, kc, lane);
        const v16h bh = load_frag(Bhi, nt0 + nt, kc, lane);
        acc[0][nt] = wmma_f16(a0, bl, acc[0][nt]);
        acc[0][nt] = wmma_f16(a0, bh, acc[0][nt]);
        acc[1][nt] = wmma_f16(a1, bl, acc[1][nt]);
        acc[1][nt] = wmma_f16(a1, bh, acc[1][nt]);
      }
    }
  }

  // Epilogue. C/D layout: VGPR r -> row r (lanes 0-15) or r+8 (lanes 16-31),
  // col = lane & 15. Add bias, apply length mask, skip padded row t==2047.
  // Skipped waves fall through with acc==0 and store zeros via the mask path.
  const int nl = lane & 15;
  const int rh = (lane >> 4) << 3;
  float bv[4];
#pragma unroll
  for (int nt = 0; nt < 4; ++nt) bv[nt] = bias[(nt0 + nt) * 16 + nl];

#pragma unroll
  for (int mt = 0; mt < 2; ++mt) {
    const int mbase = (mt0 + mt) * 16;
#pragma unroll
    for (int r = 0; r < 8; ++r) {
      const int m = mbase + r + rh;
      const int t = m & (TLEN - 1);
      if (t < TM1) {
        const bool valid = (t < limit);
        float* orow = out + ((size_t)bb * TM1 + t) * VOC + nl;
#pragma unroll
        for (int nt = 0; nt < 4; ++nt) {
          const float val = valid ? (acc[mt][nt][r] + bv[nt]) : 0.0f;
          orow[(nt0 + nt) * 16] = val;
        }
      }
    }
  }
}

// ---------------------------------------------------------------------------
extern "C" void kernel_launch(void* const* d_in, const int* in_sizes, int n_in,
                              void* d_out, int out_size, void* d_ws, size_t ws_size,
                              hipStream_t stream) {
  (void)in_sizes; (void)n_in; (void)out_size; (void)ws_size;
  // inputs: 0=times (unused), 1=codes, 2=lengths, 3=W, 4=b
  const float* codes   = (const float*)d_in[1];
  const int*   lengths = (const int*)d_in[2];
  const float* W       = (const float*)d_in[3];
  const float* bias    = (const float*)d_in[4];
  float* out = (float*)d_out;

  // Workspace layout: A fragments (128 MB) | Whi (512 KB) | Wlo (512 KB)
  _Float16* A = (_Float16*)d_ws;
  const size_t A_halves = (size_t)BATCH * TLEN * VOC;
  _Float16* Whi = A + A_halves;
  _Float16* Wlo = Whi + (size_t)VOC * VOC;

  wswz_kernel<<<dim3((VOC * VOC) / 256), dim3(256), 0, stream>>>(W, Whi, Wlo);
  prefix_kernel<<<dim3(BATCH, VOC / 256), dim3(256), 0, stream>>>(codes, lengths, A);
  gemm_kernel<<<dim3((BATCH * TLEN) / 128, VOC / 128), dim3(256), 0, stream>>>(
      A, Whi, Wlo, bias, lengths, out);
}
